// MaskSwinTransformerBlock_43258910605748
// MI455X (gfx1250) — matrix operand order repaired and checked by reading, not compiled, not run on hardware
//
#include <hip/hip_runtime.h>

typedef _Float16 f16;
typedef __attribute__((ext_vector_type(16))) _Float16 v16h;
typedef __attribute__((ext_vector_type(8)))  float    v8f;

#define NTOK   192
#define NQt    64
#define DIMC   120
#define LDC    128
#define NHEAD  6
#define HDIM   20
#define SCALEQ 0.2236067977499790f   // 1/sqrt(20)

// ---- LDS region offsets (bytes) ----
// sS (f32 [64][192]) aliases sX: X is dead after the QKV GEMMs.
// sMR/sRT alias sW: weights idle during the head loop.
#define OFF_X    0         // f16 [192][128] row-major (A operand)
#define OFF_S    0         // f32 [64][192]  (aliases sX)
#define OFF_W    49152     // f16 swizzled B-frags: 32 tiles * 32 lanes * 16 halves
#define OFF_MR   49152     // u16 [64][192] packed rel_idx|maskbit (aliases sW)
#define OFF_RT   73728     // f16 [675*6] rpb table (aliases sW tail)
#define OFF_QF   81920     // f16 swizzled A-frags: 6 heads * 4 mtiles * 32 * 16
#define OFF_KF   106496    // f16 swizzled B-frags: 6 heads * 12 ntiles * 32 * 16
#define OFF_VF   180224    // f16 swizzled B-frags: 6 heads * 6 ktiles * 2 ntl * 32 * 16
#define OFF_P    253952    // f16 [64][192] row-major (A operand)
#define OFF_O    278528    // f16 [64][128] row-major (A operand)
#define OFF_PR   294912    // f32 [64] pred mask
#define OFF_B    295168    // f32 [480] biases: qkv_b(360) + proj_b(120)
#define LDS_BYTES 297216

// A 16x32 fragment from row-major LDS: one contiguous 16-half run per lane
// (k = 16*hh + s; lowers to 2x ds_load_b128 when k0 is 8-half aligned).
static __device__ __forceinline__ v16h ldsA(const f16* base, int ld, int m0, int k0, int lane) {
  const int m = m0 + (lane & 15);
  const int hh = lane >> 4;
  v16h a;
#pragma unroll
  for (int j = 0; j < 8; ++j) {
    const int kl = (hh << 4) + (j << 1);
    a[2 * j]     = base[m * ld + k0 + kl];
    a[2 * j + 1] = base[m * ld + k0 + kl + 1];
  }
  return a;
}

// Swizzled fragment load: one 32-byte vector load per lane (2x ds_load_b128).
static __device__ __forceinline__ v16h ldsF(const f16* base, int tile, int lane) {
  return *(const v16h*)(base + (((tile << 5) + lane) << 4));
}

#define WMMA_F16(acc, a, b) \
  __builtin_amdgcn_wmma_f32_16x16x32_f16(false, (a), false, (b), (short)0, (acc), false, false)

__global__ __launch_bounds__(256)
void mask_swin_block_kernel(const float* __restrict__ x,
                            const float* __restrict__ quary0,
                            const float* __restrict__ quary1,
                            const float* __restrict__ qkv_w,
                            const float* __restrict__ qkv_b,
                            const float* __restrict__ proj_w,
                            const float* __restrict__ proj_b,
                            const float* __restrict__ rpb_table,
                            const float* __restrict__ pm_w,
                            const float* __restrict__ pm_b,
                            const float* __restrict__ attn_mask,
                            const int*   __restrict__ rel_idx,
                            float* __restrict__ out) {
  extern __shared__ char smem[];
  f16*   sX  = (f16*)(smem + OFF_X);
  float* sS  = (float*)(smem + OFF_S);
  f16*   sW  = (f16*)(smem + OFF_W);
  unsigned short* sMR = (unsigned short*)(smem + OFF_MR);
  f16*   sRT = (f16*)(smem + OFF_RT);
  f16*   sQf = (f16*)(smem + OFF_QF);
  f16*   sKf = (f16*)(smem + OFF_KF);
  f16*   sVf = (f16*)(smem + OFF_VF);
  f16*   sP  = (f16*)(smem + OFF_P);
  f16*   sO  = (f16*)(smem + OFF_O);
  float* sPr = (float*)(smem + OFF_PR);
  float* sB  = (float*)(smem + OFF_B);

  const int tid  = threadIdx.x;
  const int lane = tid & 31;
  const int wv   = tid >> 5;
  const int wb   = blockIdx.x;        // window id, 0..1023
  const int wr   = wb >> 5;           // window row (h)
  const int wc   = wb & 31;           // window col (w)

  // ---- Stage 1: gather shifted window tokens -> sX (float4 vectorized);
  //      zero pads/accumulators; stage biases ----
  for (int idx = tid; idx < NTOK * 30; idx += 256) {
    const int n = idx / 30, c4 = (idx % 30) * 4;
    const int fr = n >> 6, r = (n >> 3) & 7, cc = n & 7;
    const int hs = (wr * 8 + r + 4) & 255;   // roll(-4) gather
    const int ws = (wc * 8 + cc + 4) & 255;
    const float4 v = *(const float4*)(x + (((size_t)fr * 256 + hs) * 256 + ws) * DIMC + c4);
    f16* d = sX + n * LDC + c4;
    d[0] = (f16)v.x; d[1] = (f16)v.y; d[2] = (f16)v.z; d[3] = (f16)v.w;
  }
  for (int idx = tid; idx < NTOK * 4; idx += 256)      // zero ch 120..127
    ((unsigned*)sX)[(idx >> 2) * 64 + 60 + (idx & 3)] = 0u;
  for (int idx = tid; idx < NQt * LDC; idx += 256) sO[idx] = (f16)0.f;
  {
    unsigned* z = (unsigned*)(smem + OFF_QF);          // sQf + sKf contiguous
    for (int idx = tid; idx < (24576 + 73728) / 4; idx += 256) z[idx] = 0u;
  }
  for (int idx = tid; idx < 480; idx += 256)
    sB[idx] = (idx < 360) ? qkv_b[idx] : proj_b[idx - 360];
  __syncthreads();

  // ---- Stage 2: pred mask (|x_last64 - quary0| . pm_w + pm_b >= 0) ----
  for (int j = 0; j < 8; ++j) {
    const int q = wv * 8 + j;
    float s = 0.f;
    for (int c = lane; c < DIMC; c += 32) {
      const float xv = (float)sX[(128 + q) * LDC + c];
      s += fabsf(xv - quary0[((size_t)wb * NQt + q) * DIMC + c]) * pm_w[c];
    }
#pragma unroll
    for (int off = 16; off; off >>= 1) s += __shfl_xor(s, off);
    if (lane == 0) sPr[q] = (s + pm_b[0] >= 0.f) ? 1.f : 0.f;
  }

  // ---- Stage 3: QKV GEMMs, weights staged swizzled; outputs scatter into
  //      per-head fragment layouts for the attention GEMMs ----
  for (int p = 0; p < 3; ++p) {
    __syncthreads();
    // Stage weights in B-fragment order: k = kt*32 + 16*hh + e is contiguous
    // per slot, so global reads stay coalesced.
    for (int idx = tid; idx < 32 * 32 * 16; idx += 256) {
      const int e = idx & 15, slot = idx >> 4;
      const int ls = slot & 31, tile = slot >> 5;
      const int kt = tile >> 3, nt = tile & 7;
      const int k = kt * 32 + ((ls >> 4) << 4) + e;
      const int n = nt * 16 + (ls & 15);
      sW[idx] = (f16)((k < DIMC && n < DIMC) ? qkv_w[((size_t)(p * DIMC + n)) * DIMC + k] : 0.f);
    }
    __syncthreads();
    if (p == 0) {
      // Q: last 64 tokens; scatter into per-head A fragments (sQf).
      for (int t = wv; t < 32; t += 8) {
        const int mt = t >> 3, nt = t & 7;
        v8f acc = {};
#pragma unroll
        for (int kt = 0; kt < 4; ++kt) {
          v16h a = ldsA(sX, LDC, 128 + mt * 16, kt * 32, lane);
          v16h b = ldsF(sW, kt * 8 + nt, lane);
          acc = WMMA_F16(acc, a, b);
        }
        const int n = nt * 16 + (lane & 15);
        if (n < DIMC) {
          const float bias = sB[n];
          const int hq = n / HDIM, dl = n % HDIM;
          const int lp = (dl >= 16) ? 16 : 0;  // A-frag K half
          const int i  = dl & 15;              // element in run
#pragma unroll
          for (int r = 0; r < 8; ++r) {
            const int m = mt * 16 + ((lane >> 4) << 3) + r;
            sQf[(((hq * 4 + (m >> 4)) << 5) + (m & 15) + lp) * 16 + i] =
                (f16)((acc[r] + bias) * SCALEQ);
          }
        }
      }
    } else if (p == 1) {
      // K: scatter into per-head score-GEMM B fragments (sKf), K-mask pre-zeroed.
      for (int t = wv; t < 96; t += 8) {
        const int mt = t >> 3, nt = t & 7;
        v8f acc = {};
#pragma unroll
        for (int kt = 0; kt < 4; ++kt) {
          v16h a = ldsA(sX, LDC, mt * 16, kt * 32, lane);
          v16h b = ldsF(sW, kt * 8 + nt, lane);
          acc = WMMA_F16(acc, a, b);
        }
        const int n = nt * 16 + (lane & 15);
        if (n < DIMC) {
          const float bias = sB[DIMC + n];
          const int hq = n / HDIM, dl = n % HDIM;
          const int hh = (dl >= 16) ? 1 : 0;
          const int e  = dl - (hh << 4);
#pragma unroll
          for (int r = 0; r < 8; ++r) {
            const int m = mt * 16 + ((lane >> 4) << 3) + r;   // token 0..191
            sKf[(((hq * 12 + (m >> 4)) << 5) + (m & 15) + (hh << 4)) * 16 + e] =
                (f16)(acc[r] + bias);
          }
        }
      }
    } else {
      // V: scatter into per-head out-GEMM B fragments (sVf).
      for (int t = wv; t < 96; t += 8) {
        const int mt = t >> 3, nt = t & 7;
        v8f acc = {};
#pragma unroll
        for (int kt = 0; kt < 4; ++kt) {
          v16h a = ldsA(sX, LDC, mt * 16, kt * 32, lane);
          v16h b = ldsF(sW, kt * 8 + nt, lane);
          acc = WMMA_F16(acc, a, b);
        }
        const int n = nt * 16 + (lane & 15);
        if (n < DIMC) {
          const float bias = sB[2 * DIMC + n];
          const int hq = n / HDIM, dl = n % HDIM;
          const int ntl = dl >> 4, nl = dl & 15;
#pragma unroll
          for (int r = 0; r < 8; ++r) {
            const int m = mt * 16 + ((lane >> 4) << 3) + r;   // token 0..191
            sVf[((((hq * 6 + (m >> 5)) * 2 + ntl) << 5) + (((m >> 4) & 1) << 4) + nl) * 16 + (m & 15)] =
                (f16)(acc[r] + bias);
          }
        }
      }
    }
  }
  __syncthreads();

  // ---- Stage 3.5: cache packed (rel_idx | maskbit) and f16 rpb table in the
  //      now-idle sW region: softmax then needs zero global loads ----
  for (int idx = tid; idx < NQt * NTOK; idx += 256) {
    const int q = idx / NTOK, k = idx - q * NTOK;
    const float mval = attn_mask[((size_t)wb * NTOK + 128 + q) * NTOK + k];
    sMR[idx] = (unsigned short)((unsigned)rel_idx[idx] | (mval < -50.f ? 0x8000u : 0u));
  }
  for (int idx = tid; idx < 675 * NHEAD; idx += 256) sRT[idx] = (f16)rpb_table[idx];
  __syncthreads();

  // ---- Stage 4: per-head attention ----
  for (int h = 0; h < NHEAD; ++h) {
    // scores S = Q_h @ K_h^T : one wmma per 16x16 tile, fragments pre-swizzled
    for (int t = wv; t < 48; t += 8) {
      const int mt = t / 12, nt = t % 12;
      v16h a = ldsF(sQf, h * 4 + mt, lane);
      v16h b = ldsF(sKf, h * 12 + nt, lane);
      v8f acc = {};
      acc = WMMA_F16(acc, a, b);
      const int n = nt * 16 + (lane & 15);
#pragma unroll
      for (int r = 0; r < 8; ++r) {
        const int m = mt * 16 + ((lane >> 4) << 3) + r;
        sS[m * NTOK + n] = acc[r];
      }
    }
    __syncthreads();
    // softmax rows with rel-pos bias + shift mask (all from LDS)
    for (int j = 0; j < 8; ++j) {
      const int q = wv * 8 + j;
      float sv[6];
#pragma unroll
      for (int i = 0; i < 6; ++i) {
        const int kk = lane + 32 * i;
        const unsigned pk = sMR[q * NTOK + kk];
        float s = sS[q * NTOK + kk] + (float)sRT[(pk & 1023u) * NHEAD + h];
        if (pk & 0x8000u) s -= 100.f;
        sv[i] = s;
      }
      float mx = sv[0];
#pragma unroll
      for (int i = 1; i < 6; ++i) mx = fmaxf(mx, sv[i]);
#pragma unroll
      for (int off = 16; off; off >>= 1) mx = fmaxf(mx, __shfl_xor(mx, off));
      float sum = 0.f;
#pragma unroll
      for (int i = 0; i < 6; ++i) { sv[i] = __expf(sv[i] - mx); sum += sv[i]; }
#pragma unroll
      for (int off = 16; off; off >>= 1) sum += __shfl_xor(sum, off);
      const float inv = 1.f / sum;
#pragma unroll
      for (int i = 0; i < 6; ++i) sP[q * NTOK + lane + 32 * i] = (f16)(sv[i] * inv);
    }
    __syncthreads();
    // out_h = P @ V_h : 8 tiles -> one per wave, 6 K-steps
    {
      const int mt = wv >> 1, ntl = wv & 1;
      const int nvalid = HDIM - ntl * 16;   // 20 or 4
      v8f acc = {};
#pragma unroll
      for (int kt = 0; kt < 6; ++kt) {
        v16h a = ldsA(sP, NTOK, mt * 16, kt * 32, lane);
        v16h b = ldsF(sVf, (h * 6 + kt) * 2 + ntl, lane);
        acc = WMMA_F16(acc, a, b);
      }
      const int nl = lane & 15;
      if (nl < nvalid) {
        const int n = h * HDIM + ntl * 16 + nl;
#pragma unroll
        for (int r = 0; r < 8; ++r) {
          const int m = mt * 16 + ((lane >> 4) << 3) + r;
          sO[m * LDC + n] = (f16)acc[r];
        }
      }
    }
    __syncthreads();
  }

  // ---- Stage 5: output projection + pred blend + window-reverse/roll scatter ----
  for (int idx = tid; idx < 32 * 32 * 16; idx += 256) {
    const int e = idx & 15, slot = idx >> 4;
    const int ls = slot & 31, tile = slot >> 5;
    const int kt = tile >> 3, nt = tile & 7;
    const int k = kt * 32 + ((ls >> 4) << 4) + e;
    const int n = nt * 16 + (ls & 15);
    sW[idx] = (f16)((k < DIMC && n < DIMC) ? proj_w[(size_t)n * DIMC + k] : 0.f);
  }
  __syncthreads();
  for (int t = wv; t < 32; t += 8) {
    const int mt = t >> 3, nt = t & 7;
    v8f acc = {};
#pragma unroll
    for (int kt = 0; kt < 4; ++kt) {
      v16h a = ldsA(sO, LDC, mt * 16, kt * 32, lane);
      v16h b = ldsF(sW, kt * 8 + nt, lane);
      acc = WMMA_F16(acc, a, b);
    }
    const int n = nt * 16 + (lane & 15);
    if (n < DIMC) {
      const float bias = sB[360 + n];
#pragma unroll
      for (int r = 0; r < 8; ++r) {
        const int m = mt * 16 + ((lane >> 4) << 3) + r;   // query token 0..63
        const float pr = sPr[m];
        const float q1 = quary1[((size_t)wb * NQt + m) * DIMC + n];
        const float val = (acc[r] + bias) * pr + q1 * (1.f - pr);
        const int rr = m >> 3, cc = m & 7;
        const int fh = (wr * 8 + rr + 4) & 255;           // roll(+4) scatter
        const int fw = (wc * 8 + cc + 4) & 255;
        out[((size_t)fh * 256 + fw) * 121 + n] = val;
      }
    }
  }
  // pred channel (c = 120)
  if (tid < NQt) {
    const int m = tid;
    const int rr = m >> 3, cc = m & 7;
    const int fh = (wr * 8 + rr + 4) & 255;
    const int fw = (wc * 8 + cc + 4) & 255;
    out[((size_t)fh * 256 + fw) * 121 + 120] = sPr[m];
  }
}

extern "C" void kernel_launch(void* const* d_in, const int* in_sizes, int n_in,
                              void* d_out, int out_size, void* d_ws, size_t ws_size,
                              hipStream_t stream) {
  (void)in_sizes; (void)n_in; (void)d_ws; (void)ws_size; (void)out_size;
  const float* x         = (const float*)d_in[0];
  const float* quary0    = (const float*)d_in[1];
  const float* quary1    = (const float*)d_in[2];
  const float* qkv_w     = (const float*)d_in[3];
  const float* qkv_b     = (const float*)d_in[4];
  const float* proj_w    = (const float*)d_in[5];
  const float* proj_b    = (const float*)d_in[6];
  const float* rpb_table = (const float*)d_in[7];
  const float* pm_w      = (const float*)d_in[8];
  const float* pm_b      = (const float*)d_in[9];
  const float* attn_mask = (const float*)d_in[10];
  const int*   rel_idx   = (const int*)d_in[11];
  float* out = (float*)d_out;

  hipFuncSetAttribute(reinterpret_cast<const void*>(mask_swin_block_kernel),
                      hipFuncAttributeMaxDynamicSharedMemorySize, LDS_BYTES);
  mask_swin_block_kernel<<<1024, 256, LDS_BYTES, stream>>>(
      x, quary0, quary1, qkv_w, qkv_b, proj_w, proj_b,
      rpb_table, pm_w, pm_b, attn_mask, rel_idx, out);
}
